// EncoderLayer_67637144977429
// MI455X (gfx1250) — matrix operand order repaired and checked by reading, not compiled, not run on hardware
//
#include <hip/hip_runtime.h>
#include <stdint.h>

// ---------------------------------------------------------------------------
// Types for CDNA5 WMMA (gfx1250, wave32)
// ---------------------------------------------------------------------------
typedef __attribute__((ext_vector_type(16))) __bf16 bf16x16;
typedef __attribute__((ext_vector_type(8)))  float  f32x8;

struct U8s { unsigned u[8]; };

static __device__ inline bf16x16 frag_from(uint4 lo, uint4 hi) {
    U8s s{{lo.x, lo.y, lo.z, lo.w, hi.x, hi.y, hi.z, hi.w}};
    return __builtin_bit_cast(bf16x16, s);
}

static __device__ inline f32x8 wmma_bf16(bf16x16 a, bf16x16 b, f32x8 c) {
    // D = A(16x32 bf16) * B(32x16 bf16) + C(16x16 f32)
    return __builtin_amdgcn_wmma_f32_16x16x32_bf16(
        /*neg_a=*/false, a, /*neg_b=*/false, b,
        /*c_mod=*/(short)0, c, /*reuse_a=*/false, /*reuse_b=*/false);
}

// fp32 -> bf16 round-to-nearest-even (raw ushort storage)
static __device__ inline unsigned short f2bf(float f) {
    unsigned u = __builtin_bit_cast(unsigned, f);
    u += 0x7FFFu + ((u >> 16) & 1u);
    return (unsigned short)(u >> 16);
}
static __device__ inline unsigned pack_bf16(float a, float b) {
    return (unsigned)f2bf(a) | ((unsigned)f2bf(b) << 16);
}

// ---------------------------------------------------------------------------
// Optional CDNA5 async global->LDS copy path (ASYNCcnt), guarded so the file
// compiles on toolchains without the builtins (falls back to sync staging).
// Builtin signature (from compiler diagnostic): param0 = int4* in AS1.
// ---------------------------------------------------------------------------
#if defined(__HIP_DEVICE_COMPILE__) && \
    __has_builtin(__builtin_amdgcn_global_load_async_to_lds_b128) && \
    __has_builtin(__builtin_amdgcn_s_wait_asynccnt)
#define HAVE_ASYNC_LDS 1
#else
#define HAVE_ASYNC_LDS 0
#endif

#if HAVE_ASYNC_LDS
typedef __attribute__((ext_vector_type(4))) int v4i;
typedef __attribute__((address_space(1))) v4i g_v4i;
typedef __attribute__((address_space(3))) v4i l_v4i;
static __device__ inline void async_copy_b128(const unsigned short* g,
                                              unsigned short* l) {
    __builtin_amdgcn_global_load_async_to_lds_b128((g_v4i*)g, (l_v4i*)l, 0, 0);
}
static __device__ inline void async_wait0() {
    __builtin_amdgcn_s_wait_asynccnt(0);
}
#endif

// ---------------------------------------------------------------------------
// Problem constants
// ---------------------------------------------------------------------------
#define BB 4
#define SS 2048
#define DD 1024
#define HH 8
#define HD 128
#define FF 4096
#define NTOK (BB * SS)          // 8192 rows

// ---------------------------------------------------------------------------
// Elementwise converts
// ---------------------------------------------------------------------------
__global__ void cvt_bf16_kernel(const float* __restrict__ in,
                                unsigned short* __restrict__ out, int n) {
    int i = blockIdx.x * 256 + threadIdx.x;
    if (i < n) out[i] = f2bf(in[i]);
}

// in: fp32 [K][N] row-major  ->  out: bf16 [N][K] (transposed)
__global__ void cvt_transpose_kernel(const float* __restrict__ in,
                                     unsigned short* __restrict__ out,
                                     int K, int N) {
    int idx = blockIdx.x * 256 + threadIdx.x;
    if (idx >= K * N) return;
    int n = idx / K;
    int k = idx - n * K;
    out[idx] = f2bf(in[(size_t)k * N + n]);
}

// ---------------------------------------------------------------------------
// Generic bf16 WMMA GEMM: C[M,N] = A[M,K] * W[K,N], W given transposed [N,K].
// Block = 256 thr (8 waves). Block tile 128x128, wave tile 32(M)x64(N).
// EPI: 0 -> fp32 store, 1 -> bf16 store, 2 -> relu + bf16 store
// ---------------------------------------------------------------------------
template <int EPI>
__global__ __launch_bounds__(256, 1)
void gemm_bf16_kernel(const unsigned short* __restrict__ A,
                      const unsigned short* __restrict__ Wt,
                      float* __restrict__ Cf,
                      unsigned short* __restrict__ Cb,
                      int M, int N, int K) {
    __shared__ unsigned short As[128 * 40]; // [row][k] stride 40 (pad)
    __shared__ unsigned short Bs[128 * 40]; // [col][k] stride 40 (pad)

    const int t    = threadIdx.x;
    const int lane = t & 31;
    const int w    = t >> 5;
    const int wm   = w >> 1;      // 0..3
    const int wn   = w & 1;       // 0..1
    const int kh   = lane >> 4;   // half-wave
    const int ln16 = lane & 15;
    const int by = blockIdx.y, bx = blockIdx.x;

    const f32x8 z8 = {0.f, 0.f, 0.f, 0.f, 0.f, 0.f, 0.f, 0.f};
    f32x8 acc[2][4];
#pragma unroll
    for (int mt = 0; mt < 2; ++mt)
#pragma unroll
        for (int nt = 0; nt < 4; ++nt) acc[mt][nt] = z8;

    for (int k0 = 0; k0 < K; k0 += 32) {
        __syncthreads();
        // Stage 128x32 panels of A and Wt (each thread: 2x uint4 per panel)
#pragma unroll
        for (int i = 0; i < 2; ++i) {
            int idx = t + i * 256;            // 0..511
            int r   = idx >> 2;               // 0..127
            int c   = (idx & 3) * 8;          // 0,8,16,24
            const unsigned short* ga = A  + (size_t)(by * 128 + r) * K + k0 + c;
            const unsigned short* gb = Wt + (size_t)(bx * 128 + r) * K + k0 + c;
#if HAVE_ASYNC_LDS
            async_copy_b128(ga, &As[r * 40 + c]);
            async_copy_b128(gb, &Bs[r * 40 + c]);
#else
            *(uint4*)(&As[r * 40 + c]) = *(const uint4*)ga;
            *(uint4*)(&Bs[r * 40 + c]) = *(const uint4*)gb;
#endif
            if (k0 + 32 < K) {           // prefetch next k-panel into GL2
                __builtin_prefetch(ga + 32, 0, 1);
                __builtin_prefetch(gb + 32, 0, 1);
            }
        }
#if HAVE_ASYNC_LDS
        async_wait0();
#endif
        __syncthreads();

        bf16x16 af[2], bfr[4];
#pragma unroll
        for (int mt = 0; mt < 2; ++mt) {
            const unsigned short* p = &As[(wm * 32 + mt * 16 + ln16) * 40 + 8 * kh];
            af[mt] = frag_from(*(const uint4*)p, *(const uint4*)(p + 16));
        }
#pragma unroll
        for (int nt = 0; nt < 4; ++nt) {
            const unsigned short* p = &Bs[(wn * 64 + nt * 16 + ln16) * 40 + 8 * kh];
            bfr[nt] = frag_from(*(const uint4*)p, *(const uint4*)(p + 16));
        }
#pragma unroll
        for (int mt = 0; mt < 2; ++mt)
#pragma unroll
            for (int nt = 0; nt < 4; ++nt)
                acc[mt][nt] = wmma_bf16(af[mt], bfr[nt], acc[mt][nt]);
    }

    // Epilogue. C layout: slot r -> row r + 8*kh, lane%16 -> column
#pragma unroll
    for (int mt = 0; mt < 2; ++mt) {
#pragma unroll
        for (int nt = 0; nt < 4; ++nt) {
#pragma unroll
            for (int r = 0; r < 8; ++r) {
                int grow = by * 128 + wm * 32 + mt * 16 + r + 8 * kh;
                int gcol = bx * 128 + wn * 64 + nt * 16 + ln16;
                float v  = acc[mt][nt][r];
                size_t o = (size_t)grow * N + gcol;
                if (EPI == 0) Cf[o] = v;
                if (EPI == 1) Cb[o] = f2bf(v);
                if (EPI == 2) Cb[o] = f2bf(fmaxf(v, 0.f));
            }
        }
    }
}

// ---------------------------------------------------------------------------
// Flash attention, transposed formulation (wave32-friendly).
//   St = K_tile(64x128) @ Q^T(128x16)   per wave (wave owns 16 query rows)
//   O^T += V^T(128x64) @ P^T(64x16)
// One block per (q-tile of 128, head, batch). LDS: K tile as-is, V tile
// transposed. Each lane owns exactly one query row -> scalar m/l state.
// qkv layout: [token][3072], head h at col h*384 (+0 q, +128 k, +256 v).
// Output: valsb [token][1024] bf16 (head h at col h*128).
// ---------------------------------------------------------------------------
__global__ __launch_bounds__(256, 1)
void attn_kernel(const unsigned short* __restrict__ qkvb,
                 unsigned short* __restrict__ valsb) {
    __shared__ unsigned short Ks[64 * 136];  // [key][d]  stride 136
    __shared__ unsigned short Vs[128 * 72];  // [d][key]  stride 72

    const int t    = threadIdx.x;
    const int lane = t & 31;
    const int w    = t >> 5;
    const int kh   = lane >> 4;
    const int ln16 = lane & 15;
    const int b = blockIdx.z, h = blockIdx.y, qt = blockIdx.x;

    const size_t qrow = (size_t)b * SS + qt * 128 + w * 16 + ln16;
    const unsigned short* qptr = qkvb + qrow * (3 * DD) + h * 384;

    // Q as B-fragments of Q^T (lane%16 = query row, d = 8*kh+s, 16+8*kh+(s-8))
    bf16x16 qf[4];
#pragma unroll
    for (int kc = 0; kc < 4; ++kc) {
        const unsigned short* p = qptr + kc * 32 + 8 * kh;
        qf[kc] = frag_from(*(const uint4*)p, *(const uint4*)(p + 16));
    }

    const f32x8 z8 = {0.f, 0.f, 0.f, 0.f, 0.f, 0.f, 0.f, 0.f};
    f32x8 ot[8];
#pragma unroll
    for (int i = 0; i < 8; ++i) ot[i] = z8;

    float mrun = -1e30f, lrun = 0.f;
    const float sc = 0.08838834764831845f; // 1/sqrt(128)

    for (int kt = 0; kt < SS / 64; ++kt) {
        __syncthreads();
        // Stage K (64x128, as-is) and V (transposed into Vs[d][key])
#pragma unroll
        for (int i = 0; i < 4; ++i) {
            int idx = t + i * 256;        // 0..1023
            int r   = idx >> 4;           // key 0..63
            int c   = (idx & 15) * 8;     // d 0..120
            size_t grow = (size_t)b * SS + kt * 64 + r;
            const unsigned short* src = qkvb + grow * (3 * DD) + h * 384;
#if HAVE_ASYNC_LDS
            async_copy_b128(src + 128 + c, &Ks[r * 136 + c]);
#else
            uint4 kv = *(const uint4*)(src + 128 + c);
            *(uint4*)(&Ks[r * 136 + c]) = kv;
#endif
            uint4 vv = *(const uint4*)(src + 256 + c);
            const unsigned short* ev = (const unsigned short*)&vv;
#pragma unroll
            for (int j = 0; j < 8; ++j) Vs[(c + j) * 72 + r] = ev[j];
            if (kt + 1 < SS / 64) {       // prefetch next key tile into GL2
                __builtin_prefetch(src + 64 * 3 * DD + 128 + c, 0, 1);
                __builtin_prefetch(src + 64 * 3 * DD + 256 + c, 0, 1);
            }
        }
#if HAVE_ASYNC_LDS
        async_wait0();
#endif
        __syncthreads();

        // ---- St = K @ Q^T  (M = keys, N = query rows) ----
        f32x8 st[4];
#pragma unroll
        for (int mt = 0; mt < 4; ++mt) st[mt] = z8;
#pragma unroll
        for (int mt = 0; mt < 4; ++mt) {
#pragma unroll
            for (int kc = 0; kc < 4; ++kc) {
                const unsigned short* p = &Ks[(mt * 16 + ln16) * 136 + kc * 32 + 8 * kh];
                bf16x16 a = frag_from(*(const uint4*)p, *(const uint4*)(p + 16));
                st[mt] = wmma_bf16(a, qf[kc], st[mt]);
            }
        }

        // ---- online softmax (per-lane: one query row) ----
        float tmax = -1e30f;
#pragma unroll
        for (int mt = 0; mt < 4; ++mt)
#pragma unroll
            for (int r = 0; r < 8; ++r) {
                float v = st[mt][r] * sc;
                st[mt][r] = v;
                tmax = fmaxf(tmax, v);
            }
        tmax = fmaxf(tmax, __shfl_xor(tmax, 16, 32));
        float mnew  = fmaxf(mrun, tmax);
        float alpha = __expf(mrun - mnew);
        float tsum  = 0.f;
#pragma unroll
        for (int mt = 0; mt < 4; ++mt)
#pragma unroll
            for (int r = 0; r < 8; ++r) {
                float p = __expf(st[mt][r] - mnew);
                st[mt][r] = p;
                tsum += p;
            }
        tsum += __shfl_xor(tsum, 16, 32);
        lrun = lrun * alpha + tsum;
        mrun = mnew;
#pragma unroll
        for (int i = 0; i < 8; ++i) ot[i] *= alpha;

        // ---- pack P to bf16 pairs (keys 2j,2j+1 of this lane's half) ----
        unsigned pk[4][4];
#pragma unroll
        for (int mt = 0; mt < 4; ++mt)
#pragma unroll
            for (int j = 0; j < 4; ++j)
                pk[mt][j] = pack_bf16(st[mt][2 * j], st[mt][2 * j + 1]);

        // ---- O^T += V^T @ P^T ----
#pragma unroll
        for (int kc2 = 0; kc2 < 2; ++kc2) {
            unsigned bu[8];
#pragma unroll
            for (int j = 0; j < 8; ++j) {
                int jj  = j & 3;
                int src = ln16 + ((j < 4) ? 0 : 16);
                unsigned v0 = (unsigned)__shfl((int)pk[2 * kc2 + 0][jj], src, 32);
                unsigned v1 = (unsigned)__shfl((int)pk[2 * kc2 + 1][jj], src, 32);
                bu[j] = kh ? v1 : v0;
            }
            U8s ub{{bu[0], bu[1], bu[2], bu[3], bu[4], bu[5], bu[6], bu[7]}};
            bf16x16 bp = __builtin_bit_cast(bf16x16, ub);
#pragma unroll
            for (int mt2 = 0; mt2 < 8; ++mt2) {
                const unsigned short* p = &Vs[(mt2 * 16 + ln16) * 72 + kc2 * 32 + 16 * kh];
                bf16x16 a = frag_from(*(const uint4*)p, *(const uint4*)(p + 8));
                ot[mt2] = wmma_bf16(a, bp, ot[mt2]);
            }
        }
    }

    // ---- write O (un-transpose via C layout: slot r -> d, lane%16 -> q) ----
    float inv = 1.f / lrun;
#pragma unroll
    for (int mt2 = 0; mt2 < 8; ++mt2) {
#pragma unroll
        for (int r = 0; r < 8; r += 2) {
            int col = h * HD + mt2 * 16 + 8 * kh + r;
            unsigned pv = pack_bf16(ot[mt2][r] * inv, ot[mt2][r + 1] * inv);
            *(unsigned*)(valsb + qrow * DD + col) = pv;
        }
    }
}

// ---------------------------------------------------------------------------
// Fused residual-add + LayerNorm (no affine), row = 1024 floats.
// Writes fp32 always; bf16 copy if outb != nullptr.
// ---------------------------------------------------------------------------
__global__ void ln_kernel(const float* __restrict__ a,
                          const float* __restrict__ b,
                          float* __restrict__ outf,
                          unsigned short* __restrict__ outb) {
    const int row = blockIdx.x, t = threadIdx.x;
    const size_t base = (size_t)row * DD + t * 4;
    float4 va = *(const float4*)(a + base);
    float4 vb = *(const float4*)(b + base);
    float4 s{va.x + vb.x, va.y + vb.y, va.z + vb.z, va.w + vb.w};
    float sum = s.x + s.y + s.z + s.w;
    float sq  = s.x * s.x + s.y * s.y + s.z * s.z + s.w * s.w;
#pragma unroll
    for (int m = 1; m < 32; m <<= 1) {
        sum += __shfl_xor(sum, m, 32);
        sq  += __shfl_xor(sq, m, 32);
    }
    __shared__ float rs[8], rq[8];
    const int w = t >> 5, lane = t & 31;
    if (lane == 0) { rs[w] = sum; rq[w] = sq; }
    __syncthreads();
    sum = 0.f; sq = 0.f;
#pragma unroll
    for (int i = 0; i < 8; ++i) { sum += rs[i]; sq += rq[i]; }
    float mean = sum * (1.f / DD);
    float var  = sq * (1.f / DD) - mean * mean;
    float inv  = rsqrtf(var + 1e-5f);
    float4 o{(s.x - mean) * inv, (s.y - mean) * inv,
             (s.z - mean) * inv, (s.w - mean) * inv};
    *(float4*)(outf + base) = o;
    if (outb) {
        uint2 p{pack_bf16(o.x, o.y), pack_bf16(o.z, o.w)};
        *(uint2*)(outb + base) = p;
    }
}

// ---------------------------------------------------------------------------
// Driver
// ---------------------------------------------------------------------------
extern "C" void kernel_launch(void* const* d_in, const int* in_sizes, int n_in,
                              void* d_out, int out_size, void* d_ws, size_t ws_size,
                              hipStream_t stream) {
    (void)in_sizes; (void)n_in; (void)out_size; (void)ws_size;
    const float* x     = (const float*)d_in[0];   // [4,2048,1024]
    const float* wqkv  = (const float*)d_in[1];   // [1024,3072]
    const float* wo    = (const float*)d_in[2];   // [1024,1024]
    const float* wff1  = (const float*)d_in[3];   // [1024,4096]
    const float* wff2  = (const float*)d_in[4];   // [4096,1024]

    uint8_t* ws = (uint8_t*)d_ws;
    const size_t MB = 1u << 20;
    unsigned short* xb     = (unsigned short*)(ws + 0);          // 16MB (reused as x1b)
    unsigned short* Wq     = (unsigned short*)(ws + 16 * MB);    // 6MB  [3072][1024]
    unsigned short* Wo     = (unsigned short*)(ws + 22 * MB);    // 2MB  [1024][1024]
    unsigned short* W1     = (unsigned short*)(ws + 24 * MB);    // 8MB  [4096][1024]
    unsigned short* W2     = (unsigned short*)(ws + 32 * MB);    // 8MB  [1024][4096]
    unsigned short* qkvb   = (unsigned short*)(ws + 40 * MB);    // 48MB [8192][3072]
    unsigned short* valsb  = (unsigned short*)(ws + 88 * MB);    // 16MB [8192][1024]
    float*          attn32 = (float*)(ws + 104 * MB);            // 32MB (reused ff2out)
    float*          x1     = (float*)(ws + 136 * MB);            // 32MB
    unsigned short* x1b    = xb;                                 // overlay (xb dead)
    unsigned short* h1b    = qkvb;                               // overlay 64MB (qkvb+valsb dead)
    float*          ff2out = attn32;                             // overlay (attn32 dead)

    // 1) activations -> bf16 ; weights -> transposed bf16
    {
        int n = NTOK * DD;
        cvt_bf16_kernel<<<(n + 255) / 256, 256, 0, stream>>>(x, xb, n);
    }
    cvt_transpose_kernel<<<(DD * 3 * DD + 255) / 256, 256, 0, stream>>>(wqkv, Wq, DD, 3 * DD);
    cvt_transpose_kernel<<<(DD * DD + 255) / 256, 256, 0, stream>>>(wo, Wo, DD, DD);
    cvt_transpose_kernel<<<(DD * FF + 255) / 256, 256, 0, stream>>>(wff1, W1, DD, FF);
    cvt_transpose_kernel<<<(FF * DD + 255) / 256, 256, 0, stream>>>(wff2, W2, FF, DD);

    // 2) qkv = x @ Wqkv  -> bf16
    gemm_bf16_kernel<1><<<dim3(3 * DD / 128, NTOK / 128), 256, 0, stream>>>(
        xb, Wq, nullptr, qkvb, NTOK, 3 * DD, DD);

    // 3) attention -> valsb (bf16)
    attn_kernel<<<dim3(SS / 128, HH, BB), 256, 0, stream>>>(qkvb, valsb);

    // 4) attn_out = vals @ Wo -> fp32
    gemm_bf16_kernel<0><<<dim3(DD / 128, NTOK / 128), 256, 0, stream>>>(
        valsb, Wo, attn32, nullptr, NTOK, DD, DD);

    // 5) x1 = LN(x + attn_out)  (fp32 + bf16)
    ln_kernel<<<NTOK, 256, 0, stream>>>(x, attn32, x1, x1b);

    // 6) h1 = relu(x1 @ ff1) -> bf16
    gemm_bf16_kernel<2><<<dim3(FF / 128, NTOK / 128), 256, 0, stream>>>(
        x1b, W1, nullptr, h1b, NTOK, FF, DD);

    // 7) ff2out = h1 @ ff2 -> fp32
    gemm_bf16_kernel<0><<<dim3(DD / 128, NTOK / 128), 256, 0, stream>>>(
        h1b, W2, ff2out, nullptr, NTOK, DD, FF);

    // 8) out = LN(x1 + ff2out)
    ln_kernel<<<NTOK, 256, 0, stream>>>(x1, ff2out, (float*)d_out, nullptr);
}
